// LocalAttentionND_39754217292518
// MI455X (gfx1250) — matrix operand order repaired and checked by reading, not compiled, not run on hardware
//
#include <hip/hip_runtime.h>
#include <hip/hip_bf16.h>

// ---------------- problem constants (from reference) ----------------
#define EMBED 256
#define NHEAD 8
#define DHEAD 32
#define KW    7
#define PADW  3
#define WIN   49
#define BB    2
#define HS    64
#define WSP   64
#define LL    (HS * WSP)     // 4096
#define NROW  (BB * LL)      // 8192 tokens

typedef __attribute__((ext_vector_type(16))) _Float16 v16h;
typedef __attribute__((ext_vector_type(8)))  _Float16 v8h;
typedef __attribute__((ext_vector_type(8)))  float    v8f;

__device__ __forceinline__ float silu_f(float v)    { return v / (1.f + expf(-v)); }
__device__ __forceinline__ float sigmoid_f(float v) { return 1.f / (1.f + expf(-v)); }

// ---------------- f32 -> f16 cast ----------------
__global__ void lan_cvt_f16(const float* __restrict__ src,
                            _Float16* __restrict__ dst, int n) {
  int i = blockIdx.x * blockDim.x + threadIdx.x;
  if (i < n) dst[i] = (_Float16)src[i];
}

// ---------------- WMMA GEMM: C[M][N] = silu(A[M][K] @ W[N][K]^T + bias) ----
// One wave32 per 16x16 output tile; K stepped by 32 via v_wmma_f32_16x16x32_f16.
// A fragment (ISA 7.12.2, 16-bit A 16x32): lane row = lane&15,
//   halves 0..7  -> K = k0 + (lane>=16 ? 8:0)  + 0..7
//   halves 8..15 -> K = k0 + 16 + (lane>=16 ? 8:0) + 0..7
// B fragment (32x16): lane col = lane&15, same K mapping as A (halves 0..7 -> k0+(lane>=16?8:0)+i, 8..15 -> k0+16+(lane>=16?8:0)+i-8)
// C/D: VGPR j -> row = j + (lane>=16 ? 8:0), col = lane&15.
__global__ void lan_wmma_gemm_silu(const _Float16* __restrict__ A,
                                   const _Float16* __restrict__ W,
                                   const float* __restrict__ bias,
                                   float* __restrict__ C,
                                   int M, int N, int Kd) {
  const int lane = threadIdx.x & 31;
  const int wave = blockIdx.x * (blockDim.x >> 5) + (threadIdx.x >> 5);
  const int tn_cnt = N >> 4;
  const int tiles  = (M >> 4) * tn_cnt;
  if (wave >= tiles) return;              // wave-uniform exit: EXEC stays all-ones
  const int tm = wave / tn_cnt;
  const int tn = wave - tm * tn_cnt;
  const int rsel = lane & 15;
  const int hi   = lane >> 4;             // 0 or 1
  const _Float16* Abase = A + (size_t)(tm * 16 + rsel) * Kd;
  const _Float16* Wbase = W + (size_t)(tn * 16 + rsel) * Kd;

  v8f acc = {0.f, 0.f, 0.f, 0.f, 0.f, 0.f, 0.f, 0.f};
  union V16 { v16h v; v8h h[2]; };

  for (int k0 = 0; k0 < Kd; k0 += 32) {
    V16 a, b;
    a.h[0] = *(const v8h*)(Abase + k0 + hi * 8);
    a.h[1] = *(const v8h*)(Abase + k0 + 16 + hi * 8);
    b.h[0] = *(const v8h*)(Wbase + k0 + hi * 8);
    b.h[1] = *(const v8h*)(Wbase + k0 + 16 + hi * 8);
    if (k0 + 32 < Kd) {                   // gfx1250 global_prefetch_b8
      __builtin_prefetch(Abase + k0 + 32, 0, 1);
      __builtin_prefetch(Wbase + k0 + 32, 0, 1);
    }
    acc = __builtin_amdgcn_wmma_f32_16x16x32_f16(
        /*neg_a=*/false, a.v, /*neg_b=*/false, b.v,
        /*c_mod=*/(short)0, acc, /*reuse_a=*/false, /*reuse_b=*/false);
  }

  const int bcol = tn * 16 + rsel;
  const float bb = bias ? bias[bcol] : 0.f;
  float* crow = C + (size_t)(tm * 16 + hi * 8) * N + bcol;
#pragma unroll
  for (int j = 0; j < 8; ++j)
    crow[(size_t)j * N] = silu_f(acc[j] + bb);
}

// ---------------- per-(token,head) RMS-norm + RoPE on q,k ----------------
// NOTE: reference applies rope over axis -2 of (B,L,H,D), i.e. pos == head index.
__global__ void lan_prep_rope(const float* __restrict__ qkv,
                              const float* __restrict__ w_qn,
                              const float* __restrict__ w_kn,
                              float* __restrict__ q_rope,
                              float* __restrict__ k_rope) {
  int idx = blockIdx.x * blockDim.x + threadIdx.x;  // over NROW*NHEAD
  if (idx >= NROW * NHEAD) return;
  const int h   = idx & 7;
  const int row = idx >> 3;
  const float* qp = qkv + (size_t)row * (3 * EMBED) + h * DHEAD;
  const float* kp = qp + EMBED;
  float q[DHEAD], k[DHEAD], sq = 0.f, sk = 0.f;
#pragma unroll
  for (int d = 0; d < DHEAD; ++d) {
    q[d] = qp[d]; k[d] = kp[d];
    sq += q[d] * q[d]; sk += k[d] * k[d];
  }
  const float iq = rsqrtf(sq * (1.f / DHEAD) + 1e-6f);
  const float ik = rsqrtf(sk * (1.f / DHEAD) + 1e-6f);
#pragma unroll
  for (int d = 0; d < DHEAD; ++d) { q[d] *= iq * w_qn[d]; k[d] *= ik * w_kn[d]; }

  float* qo = q_rope + (size_t)row * EMBED + h * DHEAD;
  float* ko = k_rope + (size_t)row * EMBED + h * DHEAD;
#pragma unroll
  for (int i = 0; i < 16; ++i) {
    const float freq = expf(-(float)i * (1.f / 16.f) * 9.210340371976184f); // ln(1e4)
    const float ang  = (float)h * freq;
    const float c = cosf(ang), s = sinf(ang);
    qo[i]      = q[i] * c - q[i + 16] * s;
    qo[i + 16] = q[i] * s + q[i + 16] * c;
    ko[i]      = k[i] * c - k[i + 16] * s;
    ko[i + 16] = k[i] * s + k[i + 16] * c;
  }
}

// ---------------- 7x7 local attention, online softmax ----------------
// One lane per (pixel, head); 8 heads of a pixel are an aligned group of 8
// lanes inside a wave32 -> shuffle-reduce for the C=256 RMS norm.
__global__ void lan_attn(const float* __restrict__ q_rope,
                         const float* __restrict__ k_rope,
                         const float* __restrict__ qkv,   // for v (post-silu)
                         const float* __restrict__ wp,    // silu(x@w_wp^T+b), (NROW,16)
                         const float* __restrict__ w_on,
                         _Float16* __restrict__ afin) {   // (out_norm + v) in f16
  int idx = blockIdx.x * blockDim.x + threadIdx.x;
  if (idx >= NROW * NHEAD) return;
  const int h   = idx & 7;
  const int pix = idx >> 3;
  const int b   = pix >> 12;          // /4096
  const int l   = pix & (LL - 1);
  const int y   = l >> 6, xq = l & 63;

  const float maxd  = 4.242640687119285f;   // sqrt(2)*3
  const float width = sigmoid_f(wp[(size_t)pix * 16 + h])     * maxd + 0.5f;
  const float sharp = sigmoid_f(wp[(size_t)pix * 16 + 8 + h]) * 9.5f + 0.5f;

  float q[DHEAD];
  const float* qp = q_rope + (size_t)pix * EMBED + h * DHEAD;
#pragma unroll
  for (int d = 0; d < DHEAD; ++d) q[d] = qp[d];

  float m = -1e30f, ssum = 0.f, acc[DHEAD];
#pragma unroll
  for (int d = 0; d < DHEAD; ++d) acc[d] = 0.f;
  const float scale = 0.17677669529663687f; // 32^-0.5

  for (int wy = 0; wy < KW; ++wy) {
    for (int wx = 0; wx < KW; ++wx) {
      const int dy = wy - PADW, dx = wx - PADW;
      const int ny = y + dy, nx = xq + dx;
      const bool ok = (ny >= 0) && (ny < HS) && (nx >= 0) && (nx < WSP);
      const int  nb = b * LL + (ok ? (ny * WSP + nx) : 0);
      const float* kp = k_rope + (size_t)nb * EMBED + h * DHEAD;
      const float* vp = qkv + (size_t)nb * (3 * EMBED) + 2 * EMBED + h * DHEAD;
      float raw = 0.f;
      if (ok) {
#pragma unroll
        for (int d = 0; d < DHEAD; ++d) raw += q[d] * kp[d];
        raw *= scale;
      }
      // zero-padded windows: raw stays 0 but still takes softmax mass (v=0)
      const float dist = sqrtf((float)(dy * dy + dx * dx));
      const float sm   = sigmoid_f((width - dist) * sharp);
      const float sc   = raw - (1.f - sm) * 10000.f;
      const float nm   = fmaxf(m, sc);
      const float corr = expf(m - nm);
      const float p    = expf(sc - nm);
      ssum = ssum * corr + p;
      if (ok) {
#pragma unroll
        for (int d = 0; d < DHEAD; ++d) acc[d] = acc[d] * corr + p * vp[d];
      } else {
#pragma unroll
        for (int d = 0; d < DHEAD; ++d) acc[d] = acc[d] * corr;
      }
      m = nm;
    }
  }

  const float inv_s = 1.f / ssum;
  float o[DHEAD], sumsq = 0.f;
#pragma unroll
  for (int d = 0; d < DHEAD; ++d) { o[d] = acc[d] * inv_s; sumsq += o[d] * o[d]; }
  // RMS over full C=256: reduce across the 8 heads of this pixel (wave32 shuffles)
  sumsq += __shfl_xor(sumsq, 1, 32);
  sumsq += __shfl_xor(sumsq, 2, 32);
  sumsq += __shfl_xor(sumsq, 4, 32);
  const float invn = rsqrtf(sumsq * (1.f / EMBED) + 1e-6f);

  const float* vself = qkv + (size_t)pix * (3 * EMBED) + 2 * EMBED + h * DHEAD;
  _Float16* ao = afin + (size_t)pix * EMBED + h * DHEAD;
#pragma unroll
  for (int d = 0; d < DHEAD; ++d)
    ao[d] = (_Float16)(o[d] * invn * w_on[h * DHEAD + d] + vself[d]);
}

// ---------------- launcher ----------------
extern "C" void kernel_launch(void* const* d_in, const int* in_sizes, int n_in,
                              void* d_out, int out_size, void* d_ws, size_t ws_size,
                              hipStream_t stream) {
  const float* x     = (const float*)d_in[0];  // (2,64,64,256)
  const float* w_qkv = (const float*)d_in[1];  // (768,256)
  const float* w_qn  = (const float*)d_in[2];  // (32)
  const float* w_kn  = (const float*)d_in[3];  // (32)
  const float* w_wp  = (const float*)d_in[4];  // (16,256)
  const float* b_wp  = (const float*)d_in[5];  // (16)
  const float* w_on  = (const float*)d_in[6];  // (256)
  const float* w_out = (const float*)d_in[7];  // (256,256)
  float* out = (float*)d_out;                  // (2,64,64,256)

  // workspace carve-up (all segments 256B-aligned sizes)
  char* ws = (char*)d_ws;
  _Float16* x_h    = (_Float16*)ws; ws += (size_t)NROW * EMBED * 2;          // 4 MB
  _Float16* wqkv_h = (_Float16*)ws; ws += (size_t)3 * EMBED * EMBED * 2;     // 384 KB
  _Float16* wwp_h  = (_Float16*)ws; ws += (size_t)2 * NHEAD * EMBED * 2;     // 8 KB
  _Float16* wout_h = (_Float16*)ws; ws += (size_t)EMBED * EMBED * 2;         // 128 KB
  float*    qkv    = (float*)ws;    ws += (size_t)NROW * 3 * EMBED * 4;      // 24 MB
  float*    wp     = (float*)ws;    ws += (size_t)NROW * 16 * 4;             // 512 KB
  float*    q_rope = (float*)ws;    ws += (size_t)NROW * EMBED * 4;          // 8 MB
  float*    k_rope = (float*)ws;    ws += (size_t)NROW * EMBED * 4;          // 8 MB
  _Float16* afin   = (_Float16*)ws; ws += (size_t)NROW * EMBED * 2;          // 4 MB

  const int TB = 256;
  // 1) casts to f16
  { int n = NROW * EMBED;       lan_cvt_f16<<<(n + TB - 1) / TB, TB, 0, stream>>>(x, x_h, n); }
  { int n = 3 * EMBED * EMBED;  lan_cvt_f16<<<(n + TB - 1) / TB, TB, 0, stream>>>(w_qkv, wqkv_h, n); }
  { int n = 2 * NHEAD * EMBED;  lan_cvt_f16<<<(n + TB - 1) / TB, TB, 0, stream>>>(w_wp, wwp_h, n); }
  { int n = EMBED * EMBED;      lan_cvt_f16<<<(n + TB - 1) / TB, TB, 0, stream>>>(w_out, wout_h, n); }

  // 2) qkv = silu(x @ w_qkv^T)   M=8192 N=768 K=256
  {
    int tiles = (NROW / 16) * (3 * EMBED / 16);
    lan_wmma_gemm_silu<<<(tiles + 7) / 8, TB, 0, stream>>>(
        x_h, wqkv_h, nullptr, qkv, NROW, 3 * EMBED, EMBED);
  }
  // 3) wp = silu(x @ w_wp^T + b)  M=8192 N=16 K=256
  {
    int tiles = (NROW / 16) * (2 * NHEAD / 16);
    lan_wmma_gemm_silu<<<(tiles + 7) / 8, TB, 0, stream>>>(
        x_h, wwp_h, b_wp, wp, NROW, 2 * NHEAD, EMBED);
  }
  // 4) per-head rms + rope
  {
    int n = NROW * NHEAD;
    lan_prep_rope<<<(n + TB - 1) / TB, TB, 0, stream>>>(qkv, w_qn, w_kn, q_rope, k_rope);
  }
  // 5) local attention + rms(C) + residual, f16 staging
  {
    int n = NROW * NHEAD;
    lan_attn<<<(n + TB - 1) / TB, TB, 0, stream>>>(q_rope, k_rope, qkv, wp, w_on, afin);
  }
  // 6) out = silu(afin @ w_out^T)  M=8192 N=256 K=256
  {
    int tiles = (NROW / 16) * (EMBED / 16);
    lan_wmma_gemm_silu<<<(tiles + 7) / 8, TB, 0, stream>>>(
        afin, wout_h, nullptr, out, NROW, EMBED, EMBED);
  }
}
